// FP_module_16320875725213
// MI455X (gfx1250) — matrix operand order repaired and checked by reading, not compiled, not run on hardware
//
#include <hip/hip_runtime.h>
#include <hip/hip_bf16.h>

// ---------------------------------------------------------------------------
// Types for CDNA5 WMMA (wave32): A/B = 16 bf16 per lane, C/D = 8 f32 per lane.
// ---------------------------------------------------------------------------
typedef __attribute__((ext_vector_type(16))) __bf16 bf16x16;
typedef __attribute__((ext_vector_type(8)))  float  v8f;

struct U32x8 { uint4 lo, hi; };

__device__ __forceinline__ bf16x16 frag_from(uint4 lo, uint4 hi) {
    union { U32x8 u; bf16x16 v; } c;
    c.u.lo = lo; c.u.hi = hi;
    return c.v;
}

__device__ __forceinline__ unsigned short f2bf(float f) {
    unsigned int u = __float_as_uint(f);
    unsigned int r = u + 0x7FFFu + ((u >> 16) & 1u);   // round-to-nearest-even
    return (unsigned short)(r >> 16);
}

// Problem constants (from reference setup_inputs)
#define BB   16
#define P1   4096
#define P2   1024
#define C1   128
#define C2   256
#define MM   (BB * P1)   // 65536 rows through all GEMMs

// ---------------------------------------------------------------------------
// Pack W [N,K] f32 (row-major, out x in) into the CDNA5 B-fragment layout:
// element index e = ((nt*KT + kt)*32 + lane)*16 + j,
//   n = nt*16 + lane%16, g = lane/16, v = j/2, half = j%2,
//   k = kt*32 + 8*g + (j%8) + 16*(j/8)     (CDNA5 ISA B 32x16 layout)
// => each lane's fragment is 32 contiguous bytes; wave load fully coalesced.
// ---------------------------------------------------------------------------
__global__ void pack_w(const float* __restrict__ W, unsigned short* __restrict__ Bp,
                       int N, int K) {
    int e = blockIdx.x * 256 + threadIdx.x;
    if (e >= N * K) return;
    int j    = e & 15;
    int t    = e >> 4;
    int lane = t & 31;
    int t2   = t >> 5;
    int KT   = K >> 5;
    int kt   = t2 % KT;
    int nt   = t2 / KT;
    int n    = nt * 16 + (lane & 15);
    int g    = lane >> 4;
    int k    = kt * 32 + 8 * g + (j & 7) + 16 * (j >> 3);
    Bp[e] = f2bf(W[(size_t)n * K + k]);
}

// ---------------------------------------------------------------------------
// Phase 1: distances -> top3 -> quirky weights -> gather/max -> concat -> bf16
// One block (256 thr) handles 256 query points of one batch.
// ---------------------------------------------------------------------------
__launch_bounds__(256)
__global__ void build_x0(const float* __restrict__ xyz1, const float* __restrict__ xyz2,
                         const float* __restrict__ pts1, const float* __restrict__ pts2,
                         unsigned short* __restrict__ X0) {
    __shared__ float sx[P2], sy[P2], sz[P2];
    __shared__ int   sidx[3][256];
    __shared__ float sw[3][256];

    int tid   = threadIdx.x;
    int b     = blockIdx.x >> 4;
    int pbase = (blockIdx.x & 15) << 8;

    const float* x2 = xyz2 + (size_t)b * P2 * 3;
    for (int j = tid; j < P2; j += 256) {
        sx[j] = x2[j * 3 + 0];
        sy[j] = x2[j * 3 + 1];
        sz[j] = x2[j * 3 + 2];
    }
    __syncthreads();

    int p = pbase + tid;
    const float* x1 = xyz1 + ((size_t)b * P1 + p) * 3;
    float px = x1[0], py = x1[1], pz = x1[2];

    float d0 = 3.4e38f, d1 = 3.4e38f, d2 = 3.4e38f;
    int   i0 = 0, i1 = 0, i2 = 0;
    for (int j = 0; j < P2; ++j) {
        float dx = px - sx[j], dy = py - sy[j], dz = pz - sz[j];
        float d = dx * dx + dy * dy + dz * dz;
        if (d < d2) {
            if (d < d1) {
                if (d < d0) { d2 = d1; i2 = i1; d1 = d0; i1 = i0; d0 = d; i0 = j; }
                else        { d2 = d1; i2 = i1; d1 = d;  i1 = j; }
            } else          { d2 = d;  i2 = j; }
        }
    }
    // faithful to reference: clip(-d, 1e-10, 1e4) then inverse-distance weights
    float f0 = fminf(fmaxf(-d0, 1e-10f), 10000.f);
    float f1 = fminf(fmaxf(-d1, 1e-10f), 10000.f);
    float f2 = fminf(fmaxf(-d2, 1e-10f), 10000.f);
    float w0 = 1.f / f0, w1 = 1.f / f1, w2 = 1.f / f2;
    float s  = w0 + w1 + w2;
    sw[0][tid] = w0 / s; sw[1][tid] = w1 / s; sw[2][tid] = w2 / s;
    sidx[0][tid] = i0; sidx[1][tid] = i1; sidx[2][tid] = i2;
    __syncthreads();

    // cooperative, coalesced gather + concat ([grp(256) | pts1(128)] per row)
    const float* p2 = pts2 + (size_t)b * P2 * C2;
    const float* p1 = pts1 + ((size_t)b * P1 + pbase) * C1;
    for (int pp = 0; pp < 256; ++pp) {
        size_t row = (size_t)b * P1 + pbase + pp;
        unsigned short* xr = X0 + row * (C1 + C2);
        int   j0 = sidx[0][pp], j1 = sidx[1][pp], j2 = sidx[2][pp];
        float a0 = sw[0][pp],   a1 = sw[1][pp],   a2 = sw[2][pp];
        float g0 = p2[(size_t)j0 * C2 + tid] * a0;
        float g1 = p2[(size_t)j1 * C2 + tid] * a1;
        float g2 = p2[(size_t)j2 * C2 + tid] * a2;
        xr[tid] = f2bf(fmaxf(g0, fmaxf(g1, g2)));
        if (tid < C1) xr[C2 + tid] = f2bf(p1[(size_t)pp * C1 + tid]);
    }
}

// ---------------------------------------------------------------------------
// GEMM  Y[M,N] = A[M,K](bf16) * W^T(packed bf16) + bias, plus per-channel
// sum / sum-of-squares for BatchNorm (training-mode, biased variance).
// 8 waves/block; each wave owns a 32x64 C tile (2x4 wmma tiles).
// Compile-time KT (=K/32) and NW (waves along N); N = NW*64.
// Explicit double-buffering: fragments for kt+1 are in flight while the
// 8 WMMAs of kt execute, so waits don't drain the load counter.
// ---------------------------------------------------------------------------
template <int KT, int NW>
__launch_bounds__(256)
__global__ void gemm_bn(const unsigned short* __restrict__ A,
                        const unsigned short* __restrict__ Bp,
                        const float* __restrict__ bias,
                        float* __restrict__ Y,
                        float* __restrict__ gsum, float* __restrict__ gsq) {
    constexpr int K = KT * 32;
    constexpr int N = NW * 64;
    constexpr int blockRows = (8 / NW) * 32;

    __shared__ float sSum[256], sSq[256];
    int tid = threadIdx.x;
    sSum[tid] = 0.f; sSq[tid] = 0.f;
    __syncthreads();

    int lane  = tid & 31, wave = tid >> 5;
    int waveN = wave % NW, waveM = wave / NW;
    int rowBase = blockIdx.x * blockRows + waveM * 32;
    int colBase = waveN * 64;
    int lm = lane & 15, g = lane >> 4;

    const v8f vzero = {0.f, 0.f, 0.f, 0.f, 0.f, 0.f, 0.f, 0.f};
    v8f acc[2][4];
#pragma unroll
    for (int i = 0; i < 2; ++i)
#pragma unroll
        for (int j = 0; j < 4; ++j) acc[i][j] = vzero;

    // Hoisted fragment base pointers (all further addressing = imm offsets).
    // A: lane lm owns row (base+lm); chunks at elem kt*32+8g and +16.
    const uint4* a0b = (const uint4*)(A + (size_t)(rowBase + lm) * K + 8 * g);
    const uint4* a1b = (const uint4*)(A + (size_t)(rowBase + 16 + lm) * K + 8 * g);
    // B: lane's 32B fragment, advancing 1024B (=64 uint4) per kt.
    const uint4* bb[4];
#pragma unroll
    for (int ns = 0; ns < 4; ++ns)
        bb[ns] = (const uint4*)(Bp + (((size_t)((colBase >> 4) + ns) * KT) * 32 + lane) * 16);

    bf16x16 aC0, aC1, bC[4];
    aC0 = frag_from(a0b[0], a0b[2]);
    aC1 = frag_from(a1b[0], a1b[2]);
#pragma unroll
    for (int ns = 0; ns < 4; ++ns) bC[ns] = frag_from(bb[ns][0], bb[ns][1]);

#pragma unroll
    for (int kt = 0; kt < KT; ++kt) {
        bf16x16 aN0, aN1, bN[4];
        if (kt + 1 < KT) {                       // issue next tile's loads first
            aN0 = frag_from(a0b[(kt + 1) * 4], a0b[(kt + 1) * 4 + 2]);
            aN1 = frag_from(a1b[(kt + 1) * 4], a1b[(kt + 1) * 4 + 2]);
#pragma unroll
            for (int ns = 0; ns < 4; ++ns)
                bN[ns] = frag_from(bb[ns][(kt + 1) * 64], bb[ns][(kt + 1) * 64 + 1]);
        }
#pragma unroll
        for (int ns = 0; ns < 4; ++ns) {
            acc[0][ns] = __builtin_amdgcn_wmma_f32_16x16x32_bf16(
                false, aC0, false, bC[ns], (short)0, acc[0][ns], false, false);
            acc[1][ns] = __builtin_amdgcn_wmma_f32_16x16x32_bf16(
                false, aC1, false, bC[ns], (short)0, acc[1][ns], false, false);
        }
        if (kt + 1 < KT) {
            aC0 = aN0; aC1 = aN1;
#pragma unroll
            for (int ns = 0; ns < 4; ++ns) bC[ns] = bN[ns];
        }
    }

    // epilogue: C layout -> row = base + 8*(lane/16) + r, col = base + lane%16
#pragma unroll
    for (int ms = 0; ms < 2; ++ms) {
#pragma unroll
        for (int ns = 0; ns < 4; ++ns) {
            int n   = colBase + ns * 16 + lm;
            float bn = bias[n];
            int row = rowBase + ms * 16 + 8 * g;
            float ls = 0.f, lq = 0.f;
#pragma unroll
            for (int r = 0; r < 8; ++r) {
                float y = acc[ms][ns][r] + bn;
                Y[(size_t)(row + r) * N + n] = y;
                ls += y; lq += y * y;
            }
            atomicAdd(&sSum[n], ls);   // ds_add_f32
            atomicAdd(&sSq[n],  lq);
        }
    }
    __syncthreads();
    if (tid < N) {
        atomicAdd(&gsum[tid], sSum[tid]);
        atomicAdd(&gsq[tid],  sSq[tid]);
    }
}

__global__ void zstats(float* __restrict__ gsum, float* __restrict__ gsq) {
    int t = threadIdx.x;
    gsum[t] = 0.f; gsq[t] = 0.f;
}

__global__ void bn_fold(const float* __restrict__ gsum, const float* __restrict__ gsq,
                        const float* __restrict__ g, const float* __restrict__ be,
                        float* __restrict__ scale, float* __restrict__ shift,
                        int N, float invM) {
    int n = threadIdx.x;
    if (n < N) {
        float mean = gsum[n] * invM;
        float var  = gsq[n] * invM - mean * mean;   // biased, matches torch BN
        float sc   = g[n] * rsqrtf(var + 1e-5f);
        scale[n] = sc;
        shift[n] = be[n] - mean * sc;
    }
}

__global__ void apply_relu_bf16(const float* __restrict__ Y,
                                const float* __restrict__ scale,
                                const float* __restrict__ shift,
                                unsigned short* __restrict__ X,
                                int N, int total4) {
    int i4 = blockIdx.x * 256 + threadIdx.x;
    if (i4 >= total4) return;
    size_t i = (size_t)i4 * 4;
    int n = (int)(i % (size_t)N);
    float4 y = *(const float4*)(Y + i);
    ushort4 o;
    o.x = f2bf(fmaxf(y.x * scale[n + 0] + shift[n + 0], 0.f));
    o.y = f2bf(fmaxf(y.y * scale[n + 1] + shift[n + 1], 0.f));
    o.z = f2bf(fmaxf(y.z * scale[n + 2] + shift[n + 2], 0.f));
    o.w = f2bf(fmaxf(y.w * scale[n + 3] + shift[n + 3], 0.f));
    *(ushort4*)(X + i) = o;
}

__global__ void apply_relu_f32(const float* __restrict__ Y,
                               const float* __restrict__ scale,
                               const float* __restrict__ shift,
                               float* __restrict__ out,
                               int N, int total4) {
    int i4 = blockIdx.x * 256 + threadIdx.x;
    if (i4 >= total4) return;
    size_t i = (size_t)i4 * 4;
    int n = (int)(i % (size_t)N);
    float4 y = *(const float4*)(Y + i);
    float4 o;
    o.x = fmaxf(y.x * scale[n + 0] + shift[n + 0], 0.f);
    o.y = fmaxf(y.y * scale[n + 1] + shift[n + 1], 0.f);
    o.z = fmaxf(y.z * scale[n + 2] + shift[n + 2], 0.f);
    o.w = fmaxf(y.w * scale[n + 3] + shift[n + 3], 0.f);
    *(float4*)(out + i) = o;
}

// ---------------------------------------------------------------------------
extern "C" void kernel_launch(void* const* d_in, const int* in_sizes, int n_in,
                              void* d_out, int out_size, void* d_ws, size_t ws_size,
                              hipStream_t stream) {
    const float* xyz1 = (const float*)d_in[0];
    const float* xyz2 = (const float*)d_in[1];
    const float* pts1 = (const float*)d_in[2];
    const float* pts2 = (const float*)d_in[3];
    const float* W1 = (const float*)d_in[4];
    const float* b1 = (const float*)d_in[5];
    const float* g1 = (const float*)d_in[6];
    const float* be1 = (const float*)d_in[7];
    const float* W2 = (const float*)d_in[8];
    const float* b2 = (const float*)d_in[9];
    const float* g2 = (const float*)d_in[10];
    const float* be2 = (const float*)d_in[11];
    const float* W3 = (const float*)d_in[12];
    const float* b3 = (const float*)d_in[13];
    const float* g3 = (const float*)d_in[14];
    const float* be3 = (const float*)d_in[15];

    char* ws = (char*)d_ws;
    size_t off = 0;
    auto alloc = [&](size_t bytes) -> void* {
        void* p = ws + off;
        off = (off + bytes + 255) & ~(size_t)255;
        return p;
    };
    unsigned short* Wb1 = (unsigned short*)alloc((size_t)256 * 384 * 2);
    unsigned short* Wb2 = (unsigned short*)alloc((size_t)256 * 256 * 2);
    unsigned short* Wb3 = (unsigned short*)alloc((size_t)128 * 256 * 2);
    float* gsum = (float*)alloc(256 * 4);
    float* gsq  = (float*)alloc(256 * 4);
    float* sc1 = (float*)alloc(256 * 4); float* sh1 = (float*)alloc(256 * 4);
    float* sc2 = (float*)alloc(256 * 4); float* sh2 = (float*)alloc(256 * 4);
    float* sc3 = (float*)alloc(256 * 4); float* sh3 = (float*)alloc(256 * 4);
    unsigned short* X0 = (unsigned short*)alloc((size_t)MM * 384 * 2);  // 50 MB
    unsigned short* Xn = (unsigned short*)alloc((size_t)MM * 256 * 2);  // 33.5 MB
    float*          Y  = (float*)alloc((size_t)MM * 256 * 4);           // 67 MB

    // weight repack (tiny, once per call)
    pack_w<<<(256 * 384 + 255) / 256, 256, 0, stream>>>(W1, Wb1, 256, 384);
    pack_w<<<(256 * 256 + 255) / 256, 256, 0, stream>>>(W2, Wb2, 256, 256);
    pack_w<<<(128 * 256 + 255) / 256, 256, 0, stream>>>(W3, Wb3, 128, 256);

    // phase 1: 3-NN interpolation + concat -> X0 bf16 [65536, 384]
    build_x0<<<256, 256, 0, stream>>>(xyz1, xyz2, pts1, pts2, X0);

    const float invM = 1.f / (float)MM;

    // layer 1: [65536,384] x [384,256]
    zstats<<<1, 256, 0, stream>>>(gsum, gsq);
    gemm_bn<12, 4><<<MM / 64, 256, 0, stream>>>(X0, Wb1, b1, Y, gsum, gsq);
    bn_fold<<<1, 256, 0, stream>>>(gsum, gsq, g1, be1, sc1, sh1, 256, invM);
    apply_relu_bf16<<<(MM * 256 / 4 + 255) / 256, 256, 0, stream>>>(Y, sc1, sh1, Xn, 256, MM * 256 / 4);

    // layer 2: [65536,256] x [256,256]
    zstats<<<1, 256, 0, stream>>>(gsum, gsq);
    gemm_bn<8, 4><<<MM / 64, 256, 0, stream>>>(Xn, Wb2, b2, Y, gsum, gsq);
    bn_fold<<<1, 256, 0, stream>>>(gsum, gsq, g2, be2, sc2, sh2, 256, invM);
    apply_relu_bf16<<<(MM * 256 / 4 + 255) / 256, 256, 0, stream>>>(Y, sc2, sh2, Xn, 256, MM * 256 / 4);

    // layer 3: [65536,256] x [256,128] -> d_out f32
    zstats<<<1, 256, 0, stream>>>(gsum, gsq);
    gemm_bn<8, 2><<<MM / 128, 256, 0, stream>>>(Xn, Wb3, b3, Y, gsum, gsq);
    bn_fold<<<1, 256, 0, stream>>>(gsum, gsq, g3, be3, sc3, sh3, 128, invM);
    apply_relu_f32<<<(MM * 128 / 4 + 255) / 256, 256, 0, stream>>>(Y, sc3, sh3, (float*)d_out, 128, MM * 128 / 4);
}
